// MultiHeadAttn_55353538510859
// MI455X (gfx1250) — matrix operand order repaired and checked by reading, not compiled, not run on hardware
//
#include <hip/hip_runtime.h>
#include <hip/hip_bf16.h>

// ---------------------------------------------------------------------------
// MI455X (gfx1250) multi-head attention forward.
// B=2, S=2048, D=768, H=12, HD=64.  All matmuls on v_wmma_f32_16x16x32_f16
// (f16 operands, f32 accumulate). Flash-style online softmax avoids the
// 402MB score matrix.  Async global->LDS (ASYNCcnt) staging where the tile is
// a straight byte copy; global_prefetch for the next tiles.
// wave32, 128-thread (4-wave) blocks.
// ---------------------------------------------------------------------------

typedef __attribute__((ext_vector_type(16))) _Float16 v16h;
typedef __attribute__((ext_vector_type(8)))  float    v8f;
typedef int v4i_ __attribute__((vector_size(16)));

union AF {
    v16h h;
    unsigned int u[8];
};

// Fixed problem dimensions.
constexpr int Sc  = 2048;   // sequence length
constexpr int Dc  = 768;    // model dim (= Kd = N of every GEMM)
constexpr int Hc  = 12;     // heads
constexpr int HDc = 64;     // head dim

// LDS padded strides (halfs). B128 async LDS writes need 16B alignment:
//   LDA*2 = 80B, LDK*2 = 144B (multiples of 16, conflict-free bank spread).
constexpr int LDA = 40;     // As   [m][k]
constexpr int LDB = 34;     // Bs   [n][k]  (uint access only -> 4B ok)
constexpr int LDK = 72;     // Kt   [key][hd]
constexpr int LDV = 66;     // Vt   [hd][key], Ps (uint access only)

#if defined(__HIP_DEVICE_COMPILE__) && __has_builtin(__builtin_amdgcn_global_load_async_to_lds_b128)
#define ASYNC_CP 1
// Builtin signature (from clang diagnostic): (v4i AS1* src, v4i AS3* dst, Ii, Ii)
#define GPTR(p) ((__attribute__((address_space(1))) v4i_*)(void*)(p))
#define LPTR(p) ((__attribute__((address_space(3))) v4i_*)(void*)(p))
#else
#define ASYNC_CP 0
#endif

__device__ __forceinline__ void wait_async0() {
#if defined(__HIP_DEVICE_COMPILE__)
#if __has_builtin(__builtin_amdgcn_s_wait_asynccnt)
    __builtin_amdgcn_s_wait_asynccnt(0);
#else
    asm volatile("s_wait_asynccnt 0" ::: "memory");
#endif
#endif
}

// WMMA f16 16x16x32 operand layouts as used here (contiguous k order, A and B match):
//  A (16x32, MxK): lane l holds row r=l&15; uint pair t -> K =
//      (l>>4)*16 + 2t
//  B (32x16, KxN): lane l holds col n=l&15; half j -> K = (l>>4)*16 + j
//  C/D (16x16 f32): lane l holds col n=l&15, VGPR i -> row m = i + 8*(l>>4)

// ---------------------------------------------------------------------------
// Tiled GEMM: C[4096,768] = A[4096,768] * W[768,768] + bias.
//  A_F16=0 : A fp32 (converted while staging);  A_F16=1 : A f16 (async copy).
//  HEAD_OUT=1: f16 head-major out [(b*12+h)*S + s]*64 + d   (n = h*64+d)
//  HEAD_OUT=0: fp32 plain out[m*768 + n]
// Block tile 64x64, 4 waves each computing 32x32 (2x2 WMMA tiles), K step 32.
// ---------------------------------------------------------------------------
template<bool A_F16, bool HEAD_OUT>
__global__ __launch_bounds__(128) void gemm_f16wmma(
    const float* __restrict__ Af32, const _Float16* __restrict__ Af16,
    const float* __restrict__ W, const float* __restrict__ bias,
    float* __restrict__ outF, _Float16* __restrict__ outH)
{
    __shared__ __align__(16) _Float16 As[64 * LDA];   // [m][k]
    __shared__ __align__(16) _Float16 Bs[64 * LDB];   // [n][k]

    const int tid  = threadIdx.x;
    const int lane = tid & 31;
    const int w    = tid >> 5;
    const int l15  = lane & 15;
    const int hi   = lane >> 4;

    const int m0 = blockIdx.x * 64;
    const int n0 = blockIdx.y * 64;
    const int waveM = (w >> 1) * 32;
    const int waveN = (w & 1) * 32;

    v8f acc[2][2] = {};

    for (int kb = 0; kb < Dc; kb += 32) {
        // ---- stage A tile (64 x 32) into LDS as f16 ----
        if (A_F16) {
#if ASYNC_CP
            // 256 b128 chunks (row = 4 chunks of 8 halfs), 2 per thread.
#pragma unroll
            for (int i = 0; i < 2; ++i) {
                int e = tid + i * 128;
                int r = e >> 2, c = (e & 3) * 8;
                __builtin_amdgcn_global_load_async_to_lds_b128(
                    GPTR(&Af16[(m0 + r) * Dc + kb + c]),
                    LPTR(&As[r * LDA + c]), 0, 0);
            }
#else
            for (int e = tid; e < 1024; e += 128) {
                int r = e >> 4, uc = (e & 15) * 2;
                *(unsigned int*)&As[r * LDA + uc] =
                    *(const unsigned int*)&Af16[(m0 + r) * Dc + kb + uc];
            }
#endif
        } else {
            for (int e = tid; e < 2048; e += 128) {
                int r = e >> 5, c = e & 31;
                As[r * LDA + c] = (_Float16)Af32[(m0 + r) * Dc + kb + c];
            }
        }
        // ---- stage B tile as Bs[n][k] (fp32 W -> f16) ----
        for (int e = tid; e < 2048; e += 128) {
            int k = e >> 6, n = e & 63;
            Bs[n * LDB + k] = (_Float16)W[(kb + k) * Dc + n0 + n];
        }
        // ---- prefetch next tiles while this step computes ----
        if (kb + 32 < Dc) {
            if (tid < 32)
                __builtin_prefetch(&W[(kb + 32 + tid) * Dc + n0], 0, 1);
            else if (!A_F16)
                __builtin_prefetch(&Af32[(m0 + (tid - 32) * 2) * Dc + kb + 32], 0, 1);
        }
        if (A_F16) wait_async0();
        __syncthreads();

        AF a[2], b[2];
#pragma unroll
        for (int i = 0; i < 2; ++i) {
            int r = waveM + 16 * i + l15;
#pragma unroll
            for (int t = 0; t < 8; ++t) {
                int kp = hi * 16 + 2 * t;
                a[i].u[t] = *(const unsigned int*)&As[r * LDA + kp];
            }
        }
#pragma unroll
        for (int j = 0; j < 2; ++j) {
            int n = waveN + 16 * j + l15;
            int kbase = hi * 16;
#pragma unroll
            for (int t = 0; t < 8; ++t)
                b[j].u[t] = *(const unsigned int*)&Bs[n * LDB + kbase + 2 * t];
        }
#pragma unroll
        for (int i = 0; i < 2; ++i)
#pragma unroll
            for (int j = 0; j < 2; ++j)
                acc[i][j] = __builtin_amdgcn_wmma_f32_16x16x32_f16(
                    false, a[i].h, false, b[j].h, (short)0, acc[i][j], false, false);
        __syncthreads();
    }

    // ---- epilogue: bias add + store (all shifts/masks, no division) ----
#pragma unroll
    for (int j = 0; j < 2; ++j) {
        int n = n0 + waveN + 16 * j + l15;
        float bn = bias[n];
#pragma unroll
        for (int i = 0; i < 2; ++i) {
#pragma unroll
            for (int iv = 0; iv < 8; ++iv) {
                int m = m0 + waveM + 16 * i + iv + 8 * hi;
                float v = acc[i][j][iv] + bn;
                if (HEAD_OUT) {
                    int bb = m >> 11, s = m & (Sc - 1);     // S = 2048
                    int h = n >> 6, d = n & 63;             // HD = 64
                    outH[((bb * Hc + h) * Sc + s) * HDc + d] = (_Float16)v;
                } else {
                    outF[m * Dc + n] = v;
                }
            }
        }
    }
}

// ---------------------------------------------------------------------------
// Flash attention: one block per (b*H+h, 64-query block). 4 waves x 16 rows.
// K tile async-copied to LDS; V tile transposed manually; online softmax.
// Q pre-scaled by 1/sqrt(64)=0.125. Output f16 [B*S, 768] (h*64+d columns).
// ---------------------------------------------------------------------------
__global__ __launch_bounds__(128) void attn_kernel(
    const _Float16* __restrict__ Qh, const _Float16* __restrict__ Kh,
    const _Float16* __restrict__ Vh, _Float16* __restrict__ Oh)
{
    __shared__ __align__(16) _Float16 Kt[64 * LDK];        // [key][hd]
    __shared__ __align__(16) _Float16 Vt[64 * LDV];        // [hd][key]
    __shared__ __align__(16) _Float16 Ps[4 * 16 * LDV];    // per-wave P scratch

    const int tid = threadIdx.x, lane = tid & 31, w = tid >> 5;
    const int l15 = lane & 15, hi = lane >> 4;
    const int qblk = blockIdx.x;
    const int bh   = blockIdx.y;              // b*12 + h

    const _Float16* Qb = Qh + bh * Sc * HDc;
    const _Float16* Kb = Kh + bh * Sc * HDc;
    const _Float16* Vb = Vh + bh * Sc * HDc;
    _Float16* Pw = &Ps[w * 16 * LDV];
    const int q0 = qblk * 64 + w * 16;

    // ---- load Q fragments once (A layout), fold in 1/sqrt(HD) ----
    AF qa[2];
#pragma unroll
    for (int ks = 0; ks < 2; ++ks) {
        int r = q0 + l15;
#pragma unroll
        for (int t = 0; t < 8; ++t) {
            int kp = ks * 32 + hi * 16 + 2 * t;
            qa[ks].u[t] = *(const unsigned int*)&Qb[r * HDc + kp];
        }
        qa[ks].h = qa[ks].h * (_Float16)0.125f;
    }

    float mi[8], li[8];
    v8f oacc[4] = {};
#pragma unroll
    for (int iv = 0; iv < 8; ++iv) { mi[iv] = -1e30f; li[iv] = 0.f; }

    for (int kb = 0; kb < Sc; kb += 64) {
        // ---- stage K tile [key][hd] : straight byte copy -> async to LDS ----
#if ASYNC_CP
#pragma unroll
        for (int i = 0; i < 4; ++i) {               // 512 b128 chunks / 128 thr
            int e = tid + i * 128;
            int key = e >> 3, c = (e & 7) * 8;
            __builtin_amdgcn_global_load_async_to_lds_b128(
                GPTR(&Kb[(kb + key) * HDc + c]),
                LPTR(&Kt[key * LDK + c]), 0, 0);
        }
#else
        for (int e = tid; e < 2048; e += 128) {
            int key = e >> 5, uc = (e & 31) * 2;
            *(unsigned int*)&Kt[key * LDK + uc] =
                *(const unsigned int*)&Kb[(kb + key) * HDc + uc];
        }
#endif
        // ---- stage V tile transposed: Vt[hd][key] (overlaps async K copy) ----
        for (int e = tid; e < 2048; e += 128) {
            int key = e >> 5, du = (e & 31) * 2;
            union { unsigned int u; _Float16 f[2]; } cv;
            cv.u = *(const unsigned int*)&Vb[(kb + key) * HDc + du];
            Vt[du * LDV + key]       = cv.f[0];
            Vt[(du + 1) * LDV + key] = cv.f[1];
        }
        // ---- prefetch next K/V tiles ----
        if (kb + 64 < Sc) {
            __builtin_prefetch(&Kb[(kb + 64) * HDc + tid * 32], 0, 1);
            __builtin_prefetch(&Vb[(kb + 64) * HDc + tid * 32], 0, 1);
        }
#if ASYNC_CP
        wait_async0();
#endif
        __syncthreads();

        // ---- S = (Q/8) K^T : 16q x 64k, 4 subtiles x 2 k-steps ----
        v8f sc[4];
#pragma unroll
        for (int j = 0; j < 4; ++j) {
            v8f s = {};
#pragma unroll
            for (int ks = 0; ks < 2; ++ks) {
                AF bf;
                int n = j * 16 + l15;
                int kbase = ks * 32 + hi * 16;
#pragma unroll
                for (int t = 0; t < 8; ++t)
                    bf.u[t] = *(const unsigned int*)&Kt[n * LDK + kbase + 2 * t];
                s = __builtin_amdgcn_wmma_f32_16x16x32_f16(
                        false, qa[ks].h, false, bf.h, (short)0, s, false, false);
            }
            sc[j] = s;
        }

        // ---- online softmax (rows spread over 16-lane halves) ----
        float nm[8], corr[8], psum[8];
#pragma unroll
        for (int iv = 0; iv < 8; ++iv) {
            float bm = fmaxf(fmaxf(sc[0][iv], sc[1][iv]), fmaxf(sc[2][iv], sc[3][iv]));
            for (int off = 1; off < 16; off <<= 1)
                bm = fmaxf(bm, __shfl_xor(bm, off, 16));
            float newm = fmaxf(mi[iv], bm);
            corr[iv] = exp2f((mi[iv] - newm) * 1.44269504f);
            nm[iv] = newm;
            psum[iv] = 0.f;
        }
#pragma unroll
        for (int j = 0; j < 4; ++j) {
#pragma unroll
            for (int iv = 0; iv < 8; ++iv) {
                float p = exp2f((sc[j][iv] - nm[iv]) * 1.44269504f);
                psum[iv] += p;
                // C-layout elem (row=iv+8*hi, col=j*16+l15) -> row-major LDS
                Pw[(iv + 8 * hi) * LDV + j * 16 + l15] = (_Float16)p;
            }
        }
#pragma unroll
        for (int iv = 0; iv < 8; ++iv) {
            float ps = psum[iv];
            for (int off = 1; off < 16; off <<= 1)
                ps += __shfl_xor(ps, off, 16);
            li[iv] = li[iv] * corr[iv] + ps;
            mi[iv] = nm[iv];
#pragma unroll
            for (int j = 0; j < 4; ++j) oacc[j][iv] *= corr[iv];
        }

        // ---- O += P V  (same-wave LDS write->read is in-order, ISA §7.3) ----
#pragma unroll
        for (int ks = 0; ks < 2; ++ks) {
            AF pa;
#pragma unroll
            for (int t = 0; t < 8; ++t) {
                int kp = ks * 32 + hi * 16 + 2 * t;
                pa.u[t] = *(const unsigned int*)&Pw[l15 * LDV + kp];
            }
#pragma unroll
            for (int j = 0; j < 4; ++j) {
                AF vb;
                int d = j * 16 + l15;
                int kbase = ks * 32 + hi * 16;
#pragma unroll
                for (int t = 0; t < 8; ++t)
                    vb.u[t] = *(const unsigned int*)&Vt[d * LDV + kbase + 2 * t];
                oacc[j] = __builtin_amdgcn_wmma_f32_16x16x32_f16(
                              false, pa.h, false, vb.h, (short)0, oacc[j], false, false);
            }
        }
        __syncthreads();
    }

    // ---- finalize: O / l, store f16 at [b*S+s][h*64+d] ----
    const int b = bh / Hc, h = bh - b * Hc;
#pragma unroll
    for (int iv = 0; iv < 8; ++iv) {
        float inv = 1.0f / li[iv];
        int srow = q0 + iv + 8 * hi;
        int mrow = b * Sc + srow;
#pragma unroll
        for (int j = 0; j < 4; ++j) {
            int d = j * 16 + l15;
            Oh[mrow * Dc + h * HDc + d] = (_Float16)(oacc[j][iv] * inv);
        }
    }
}

// ---------------------------------------------------------------------------
extern "C" void kernel_launch(void* const* d_in, const int* in_sizes, int n_in,
                              void* d_out, int out_size, void* d_ws, size_t ws_size,
                              hipStream_t stream) {
    (void)in_sizes; (void)n_in; (void)out_size; (void)ws_size;
    const float* Xq = (const float*)d_in[0];
    const float* Xk = (const float*)d_in[1];
    const float* Xv = (const float*)d_in[2];
    const float* Wq = (const float*)d_in[3];
    const float* bq = (const float*)d_in[4];
    const float* Wk = (const float*)d_in[5];
    const float* bk = (const float*)d_in[6];
    const float* Wv = (const float*)d_in[7];
    const float* bv = (const float*)d_in[8];
    const float* Wo = (const float*)d_in[9];
    const float* bo = (const float*)d_in[10];
    float* out = (float*)d_out;

    const int M = 2 * Sc;                           // 4096
    const size_t ph = (size_t)2 * Hc * Sc * HDc;    // 3,145,728 halfs

    _Float16* Qh = (_Float16*)d_ws;
    _Float16* Kh = Qh + ph;
    _Float16* Vh = Kh + ph;
    _Float16* Oh = Vh + ph;                         // 4 x 6MB = 24MB of ws

    dim3 blk(128);
    dim3 gG(M / 64, Dc / 64);                       // (64, 12)

    gemm_f16wmma<false, true><<<gG, blk, 0, stream>>>(Xq, nullptr, Wq, bq, nullptr, Qh);
    gemm_f16wmma<false, true><<<gG, blk, 0, stream>>>(Xk, nullptr, Wk, bk, nullptr, Kh);
    gemm_f16wmma<false, true><<<gG, blk, 0, stream>>>(Xv, nullptr, Wv, bv, nullptr, Vh);

    attn_kernel<<<dim3(Sc / 64, 2 * Hc), blk, 0, stream>>>(Qh, Kh, Vh, Oh);

    gemm_f16wmma<true, false><<<gG, blk, 0, stream>>>(nullptr, Oh, Wo, bo, out, nullptr);
}
